// LSTM_8349416423854
// MI455X (gfx1250) — compile-verified, matrix operand order checked by reading, not run on hardware
//
#include <hip/hip_runtime.h>
#include <hip/hip_bf16.h>

typedef __bf16 bf16_t;
typedef __attribute__((ext_vector_type(16))) __bf16 v16bf;
typedef __attribute__((ext_vector_type(8)))  __bf16 v8bf;
typedef __attribute__((ext_vector_type(4)))  __bf16 v4bf;
typedef __attribute__((ext_vector_type(8)))  float  v8f;
typedef __attribute__((ext_vector_type(4)))  unsigned int v4u;
typedef __attribute__((ext_vector_type(8)))  int v8i;
typedef __attribute__((ext_vector_type(4)))  int v4i;

#define N_BATCH 128
#define T_SEQ   512
#define I_DIM   1024
#define H_DIM   1024
#define G4      4096   // 4*H

// ---------------------------------------------------------------------------
// TDM: DMA a 128-row x 32-col bf16 tile (row stride 1024 elements) from global
// memory into LDS at byte offset lds_byte_off. 2D tensor descriptor per ISA §8.
// ---------------------------------------------------------------------------
__device__ __forceinline__ void tdm_load_tile_128x32(const bf16_t* gsrc,
                                                     unsigned int lds_byte_off,
                                                     unsigned int tensor_rows) {
  unsigned long long ga = (unsigned long long)(const void*)gsrc;
  v4u g0;
  g0[0] = 1u;                                                      // count=1 (valid)
  g0[1] = lds_byte_off;                                            // D#.lds_addr
  g0[2] = (unsigned int)ga;                                        // global_addr[31:0]
  g0[3] = (unsigned int)((ga >> 32) & 0x1FFFFFFull) | (2u << 30);  // addr[56:32] | type=2
  v8i g1;
  g1[0] = (int)(1u << 16);                          // data_size=1 (2-byte elements)
  g1[1] = (int)(1024u << 16);                       // tensor_dim0 = 1024 (bits 79:48)
  g1[2] = (int)((tensor_rows & 0xFFFFu) << 16);     // tensor_dim1[15:0]  (bits 111:80)
  g1[3] = (int)((tensor_rows >> 16) | (32u << 16)); // tensor_dim1[31:16] | tile_dim0=32
  g1[4] = 128;                                      // tile_dim1=128, tile_dim2=0
  g1[5] = 1024;                                     // tensor_dim0_stride low32 = 1024
  g1[6] = 0;
  g1[7] = 0;
  v4i gz4 = {0, 0, 0, 0};
  v8i gz8 = {0, 0, 0, 0, 0, 0, 0, 0};
  __builtin_amdgcn_tensor_load_to_lds(g0, g1, gz4, gz4, gz8, 0);
}

// Escape the LDS array's address so alias analysis must assume the TDM engine
// writes it (the descriptor only carries the LDS offset as an integer).
__device__ __forceinline__ void lds_escape(const void* p) {
  asm volatile("" :: "s"(p) : "memory");
}
__device__ __forceinline__ void compiler_mem_barrier() {
  asm volatile("" ::: "memory");
}

// ---------------------------------------------------------------------------
// Pack a [1024 x 4096] weight (concat of 4 gate matrices along columns) from
// fp32 row-major into bf16 WMMA B-fragment-major layout:
//   packed[((kt*256 + ntile)*32 + lane)*16 + e]
// ---------------------------------------------------------------------------
__global__ void pack_w_kernel(const float* __restrict__ w0, const float* __restrict__ w1,
                              const float* __restrict__ w2, const float* __restrict__ w3,
                              bf16_t* __restrict__ packed) {
  const int total = I_DIM * G4;   // 4,194,304
  for (int idx = blockIdx.x * blockDim.x + threadIdx.x; idx < total;
       idx += gridDim.x * blockDim.x) {
    int e    = idx & 15;
    int lane = (idx >> 4) & 31;
    int frag = idx >> 9;
    int ntile = frag & 255;       // 4096/16 column tiles
    int kt    = frag >> 8;        // K tile of 32
    int lo = lane & 15, hi = lane >> 4;
    int kLocal = (e < 8) ? (hi * 8 + e) : (hi * 8 + e + 8);
    int k = kt * 32 + kLocal;
    int n = ntile * 16 + lo;
    int gate = n >> 10;
    int c = n & 1023;
    const float* w = (gate == 0) ? w0 : (gate == 1) ? w1 : (gate == 2) ? w2 : w3;
    packed[idx] = (bf16_t)w[(size_t)k * H_DIM + c];
  }
}

__global__ void bias_kernel(const float* bii, const float* bhi, const float* bif,
                            const float* bhf, const float* big, const float* bhg,
                            const float* bio, const float* bho, float* __restrict__ b) {
  int idx = blockIdx.x * blockDim.x + threadIdx.x;
  if (idx >= G4) return;
  int gate = idx >> 10, c = idx & 1023;
  const float* bi = (gate == 0) ? bii : (gate == 1) ? bif : (gate == 2) ? big : bio;
  const float* bh = (gate == 0) ? bhi : (gate == 1) ? bhf : (gate == 2) ? bhg : bho;
  b[idx] = bi[c] + bh[c];
}

__global__ void cvt_x_kernel(const float* __restrict__ x, bf16_t* __restrict__ xb) {
  size_t idx = (size_t)blockIdx.x * blockDim.x + threadIdx.x;   // one float4 each
  const float4 v = ((const float4*)x)[idx];
  v4bf o;
  o[0] = (bf16_t)v.x; o[1] = (bf16_t)v.y; o[2] = (bf16_t)v.z; o[3] = (bf16_t)v.w;
  *(v4bf*)&xb[idx * 4] = o;
}

// ---------------------------------------------------------------------------
// Phase 1: Xg[65536,4096] = Xb[65536,1024] @ WiP + bias   (bf16 WMMA, fp32 acc)
// Block tile 128x128, 8 waves. A tile staged into double-buffered LDS by TDM.
// ---------------------------------------------------------------------------
__global__ __launch_bounds__(256, 2)
void gemm_xg_kernel(const bf16_t* __restrict__ Xb, const bf16_t* __restrict__ Wp,
                    const float* __restrict__ bias, float* __restrict__ Xg) {
  __shared__ __align__(128) bf16_t ldsA[2][128 * 32];   // 2 x 8 KB, group offset 0
  const int tid  = threadIdx.x;
  const int wave = tid >> 5;
  const int lane = tid & 31;
  const int lo = lane & 15, hi = lane >> 4;
  const int mBase = blockIdx.y * 128;
  const int nBase = blockIdx.x * 128;

  lds_escape(&ldsA[0][0]);   // TDM writes this array behind the compiler's back

  v8f acc[8];
#pragma unroll
  for (int i = 0; i < 8; ++i) acc[i] = v8f{};

  const v16bf* Bp = (const v16bf*)Wp;
  const bf16_t* Abase = Xb + (size_t)mBase * I_DIM;

  if (tid < 32)
    tdm_load_tile_128x32(Abase, 0u, (unsigned)(N_BATCH * T_SEQ));

  for (int kt = 0; kt < 32; ++kt) {
    if (tid < 32) {
      if (kt < 31) {
        tdm_load_tile_128x32(Abase + (kt + 1) * 32, ((kt + 1) & 1) * 8192u,
                             (unsigned)(N_BATCH * T_SEQ));
        __builtin_amdgcn_s_wait_tensorcnt(1);   // current buffer's DMA complete
      } else {
        __builtin_amdgcn_s_wait_tensorcnt(0);
      }
    }
    __syncthreads();
    compiler_mem_barrier();                     // LDS contents changed by TDM
    const bf16_t* buf = ldsA[kt & 1];
    const int mLocal = wave * 16 + lo;
    v8bf a0 = *(const v8bf*)&buf[mLocal * 32 + hi * 8];
    v8bf a1 = *(const v8bf*)&buf[mLocal * 32 + 16 + hi * 8];
    v16bf a = __builtin_shufflevector(a0, a1, 0, 1, 2, 3, 4, 5, 6, 7,
                                      8, 9, 10, 11, 12, 13, 14, 15);
#pragma unroll
    for (int nt = 0; nt < 8; ++nt) {
      size_t f = (size_t)kt * 256 + (nBase >> 4) + nt;
      v16bf b = Bp[f * 32 + lane];
      acc[nt] = __builtin_amdgcn_wmma_f32_16x16x32_bf16(
          false, a, false, b, (short)0, acc[nt], false, false);
    }
    __syncthreads();   // protect buffer reuse two iterations later
  }

#pragma unroll
  for (int nt = 0; nt < 8; ++nt) {
    const int col = nBase + nt * 16 + lo;
    const float bcol = bias[col];
#pragma unroll
    for (int r = 0; r < 8; ++r) {
      const int row = mBase + wave * 16 + hi * 8 + r;
      Xg[(size_t)row * G4 + col] = acc[nt][r] + bcol;
    }
  }
}

// ---------------------------------------------------------------------------
// Phase 2: one recurrent step. G = Xg[:,t,:] + h @ WhP, fused gates.
// Each block owns 32 columns of EVERY gate (frags 0,1=i 2,3=f 4,5=g 6,7=o),
// so i/f/g/o for a given (n,col) sit in the same lane element -> gates fuse.
// ---------------------------------------------------------------------------
__global__ __launch_bounds__(256, 2)
void lstm_step_kernel(const bf16_t* __restrict__ Hb, const bf16_t* __restrict__ Wp,
                      const float* __restrict__ Xg, float* __restrict__ Hout,
                      float* __restrict__ Cst, bf16_t* __restrict__ HbOut, int t) {
  __shared__ __align__(128) bf16_t ldsA[2][128 * 32];
  const int tid  = threadIdx.x;
  const int wave = tid >> 5;
  const int lane = tid & 31;
  const int lo = lane & 15, hi = lane >> 4;
  const int j0 = blockIdx.x * 32;   // column base within each gate

  v8f acc[8];
#pragma unroll
  for (int i = 0; i < 8; ++i) acc[i] = v8f{};

  if (t > 0) {
    lds_escape(&ldsA[0][0]);
    const v16bf* Bp = (const v16bf*)Wp;
    if (tid < 32)
      tdm_load_tile_128x32(Hb, 0u, (unsigned)N_BATCH);
    for (int kt = 0; kt < 32; ++kt) {
      if (tid < 32) {
        if (kt < 31) {
          tdm_load_tile_128x32(Hb + (kt + 1) * 32, ((kt + 1) & 1) * 8192u,
                               (unsigned)N_BATCH);
          __builtin_amdgcn_s_wait_tensorcnt(1);
        } else {
          __builtin_amdgcn_s_wait_tensorcnt(0);
        }
      }
      __syncthreads();
      compiler_mem_barrier();
      const bf16_t* buf = ldsA[kt & 1];
      const int mLocal = wave * 16 + lo;
      v8bf a0 = *(const v8bf*)&buf[mLocal * 32 + hi * 8];
      v8bf a1 = *(const v8bf*)&buf[mLocal * 32 + 16 + hi * 8];
      v16bf a = __builtin_shufflevector(a0, a1, 0, 1, 2, 3, 4, 5, 6, 7,
                                        8, 9, 10, 11, 12, 13, 14, 15);
#pragma unroll
      for (int nt = 0; nt < 8; ++nt) {
        const int gate = nt >> 1, p = nt & 1;
        const int ntile = gate * 64 + (j0 >> 4) + p;
        size_t f = (size_t)kt * 256 + ntile;
        v16bf b = Bp[f * 32 + lane];
        acc[nt] = __builtin_amdgcn_wmma_f32_16x16x32_bf16(
            false, a, false, b, (short)0, acc[nt], false, false);
      }
      __syncthreads();
    }
  }

  // fused gates + state update; write h (fp32 + bf16) and c
#pragma unroll
  for (int p = 0; p < 2; ++p) {
    const int col = j0 + p * 16 + lo;
#pragma unroll
    for (int r = 0; r < 8; ++r) {
      const int n = wave * 16 + hi * 8 + r;             // batch row
      const size_t xrow = ((size_t)n * T_SEQ + t) * G4;
      float xi = acc[0 + p][r] + Xg[xrow + col];
      float xf = acc[2 + p][r] + Xg[xrow + 1024 + col];
      float xg = acc[4 + p][r] + Xg[xrow + 2048 + col];
      float xo = acc[6 + p][r] + Xg[xrow + 3072 + col];
      float ig = 1.0f / (1.0f + __expf(-xi));
      float fg = 1.0f / (1.0f + __expf(-xf));
      float gg = tanhf(xg);
      float og = 1.0f / (1.0f + __expf(-xo));
      float cold = (t > 0) ? Cst[(size_t)n * H_DIM + col] : 0.0f;
      float cnew = fg * cold + ig * gg;
      float hnew = og * tanhf(cnew);
      Cst[(size_t)n * H_DIM + col]  = cnew;
      Hout[(size_t)n * H_DIM + col] = hnew;
      HbOut[(size_t)n * H_DIM + col] = (bf16_t)hnew;
    }
  }
}

// ---------------------------------------------------------------------------
extern "C" void kernel_launch(void* const* d_in, const int* in_sizes, int n_in,
                              void* d_out, int out_size, void* d_ws, size_t ws_size,
                              hipStream_t stream) {
  (void)in_sizes; (void)n_in; (void)out_size; (void)ws_size;
  const float* x   = (const float*)d_in[0];
  const float* wii = (const float*)d_in[1];
  const float* whi = (const float*)d_in[2];
  const float* wif = (const float*)d_in[3];
  const float* whf = (const float*)d_in[4];
  const float* wig = (const float*)d_in[5];
  const float* whg = (const float*)d_in[6];
  const float* wio = (const float*)d_in[7];
  const float* who = (const float*)d_in[8];
  const float* bii = (const float*)d_in[9];
  const float* bhi = (const float*)d_in[10];
  const float* bif = (const float*)d_in[11];
  const float* bhf = (const float*)d_in[12];
  const float* big = (const float*)d_in[13];
  const float* bhg = (const float*)d_in[14];
  const float* bio = (const float*)d_in[15];
  const float* bho = (const float*)d_in[16];

  char* ws = (char*)d_ws;
  size_t off = 0;
  bf16_t* WiP = (bf16_t*)(ws + off); off += (size_t)I_DIM * G4 * 2;           // 8 MB
  bf16_t* WhP = (bf16_t*)(ws + off); off += (size_t)H_DIM * G4 * 2;           // 8 MB
  float*  bsum = (float*)(ws + off); off += 64 * 1024;                        // 16 KB (+pad)
  bf16_t* Xb  = (bf16_t*)(ws + off); off += (size_t)N_BATCH * T_SEQ * I_DIM * 2; // 128 MB
  float*  Xg  = (float*) (ws + off); off += (size_t)N_BATCH * T_SEQ * G4 * 4;    // 1 GB
  bf16_t* H0  = (bf16_t*)(ws + off); off += (size_t)N_BATCH * H_DIM * 2;
  bf16_t* H1  = (bf16_t*)(ws + off); off += (size_t)N_BATCH * H_DIM * 2;

  float* Hout = (float*)d_out;                               // [128,1024] h_T
  float* Cst  = ((float*)d_out) + (size_t)N_BATCH * H_DIM;   // [128,1024] c_T

  pack_w_kernel<<<4096, 256, 0, stream>>>(wii, wif, wig, wio, WiP);
  pack_w_kernel<<<4096, 256, 0, stream>>>(whi, whf, whg, who, WhP);
  bias_kernel<<<16, 256, 0, stream>>>(bii, bhi, bif, bhf, big, bhg, bio, bho, bsum);
  cvt_x_kernel<<<65536, 256, 0, stream>>>(x, Xb);      // 64M floats, 4/thread

  dim3 g1(32, 512);                                    // 4096/128 x 65536/128
  gemm_xg_kernel<<<g1, 256, 0, stream>>>(Xb, WiP, bsum, Xg);

  for (int t = 0; t < T_SEQ; ++t) {
    const bf16_t* hin = (t & 1) ? H1 : H0;
    bf16_t* hout      = (t & 1) ? H0 : H1;
    lstm_step_kernel<<<32, 256, 0, stream>>>(hin, WhP, Xg, Hout, Cst, hout, t);
  }
}
